// MOELinearB_63161789055646
// MI455X (gfx1250) — compile-verified, matrix operand order checked by reading, not run on hardware
//
#include <hip/hip_runtime.h>

typedef __attribute__((ext_vector_type(16))) __bf16 v16bf;
typedef __attribute__((ext_vector_type(8)))  __bf16 v8bf;
typedef __attribute__((ext_vector_type(4)))  __bf16 v4bf;
typedef __attribute__((ext_vector_type(8)))  float  v8f;

#define NUM_E   8
#define NROWS   4096     // N (rows of x per expert)
#define RDIM    128      // K
#define ODIM    4096     // out features
#define KC      64       // K chunk staged in LDS
#define PAD     8        // bf16 elements of row padding (16 B) -> conflict-free strides
#define LDSROW  (KC + PAD)

// Build a 16x32 (or 32x16) bf16 WMMA fragment from one LDS row.
// Per CDNA5 ISA layout: lanes 0-15 hold K runs {kb+0..7, kb+16..23},
// lanes 16-31 hold {kb+8..15, kb+24..31}; each run is one 16-byte read.
__device__ inline v16bf load_frag(const __bf16* __restrict__ rowptr, int kb, int half) {
    const v8bf r0 = *(const v8bf*)(rowptr + kb + half * 8);
    const v8bf r1 = *(const v8bf*)(rowptr + kb + 16 + half * 8);
    return __builtin_shufflevector(r0, r1, 0, 1, 2, 3, 4, 5, 6, 7,
                                           8, 9, 10, 11, 12, 13, 14, 15);
}

__device__ inline void split_bf16(float x, __bf16& h, __bf16& l) {
    h = (__bf16)x;                    // RNE high part
    l = (__bf16)(x - (float)h);       // residual
}

__global__ __launch_bounds__(256, 1)
void moe_grouped_gemm_bf16x3(const float* __restrict__ X,
                             const float* __restrict__ W,
                             float* __restrict__ out)
{
    __shared__ __bf16 a_hi[128][LDSROW];
    __shared__ __bf16 a_lo[128][LDSROW];
    __shared__ __bf16 b_hi[128][LDSROW];
    __shared__ __bf16 b_lo[128][LDSROW];

    const int tid  = threadIdx.x;
    const int lane = tid & 31;
    const int wave = tid >> 5;
    const int wr   = wave >> 2;   // 0..1 : 64-row band of the 128x128 tile
    const int wc   = wave & 3;    // 0..3 : 32-col band
    const int half = lane >> 4;   // K-run selector per ISA layout
    const int lr   = lane & 15;   // row (A) / col (B) within 16

    const int e     = blockIdx.z;
    const int tileM = blockIdx.y;
    const int tileN = blockIdx.x;

    const float* __restrict__ Abase = X + ((size_t)e * NROWS + (size_t)tileM * 128) * RDIM;
    const float* __restrict__ Bbase = W + ((size_t)e * ODIM  + (size_t)tileN * 128) * RDIM;

    v8f acc[4][2];
#pragma unroll
    for (int i = 0; i < 4; ++i)
#pragma unroll
        for (int j = 0; j < 2; ++j)
#pragma unroll
            for (int v = 0; v < 8; ++v)
                acc[i][j][v] = 0.0f;

    const int rowbase = wr * 64;
    const int colbase = wc * 32;

    for (int kc = 0; kc < RDIM; kc += KC) {
        // ---- stage 128x64 f32 tiles of x and W into LDS as bf16 hi/lo ----
#pragma unroll
        for (int it = 0; it < 8; ++it) {
            const int idx = tid + it * 256;      // 2048 float4 slots per tile
            const int row = idx >> 4;            // 0..127
            const int c4  = (idx & 15) * 4;      // 0..60 step 4

            const float4 av = *(const float4*)(Abase + (size_t)row * RDIM + kc + c4);
            const float4 bv = *(const float4*)(Bbase + (size_t)row * RDIM + kc + c4);

            const float af[4] = {av.x, av.y, av.z, av.w};
            const float bf[4] = {bv.x, bv.y, bv.z, bv.w};
            v4bf ah, al, bh, bl;
#pragma unroll
            for (int q = 0; q < 4; ++q) {
                __bf16 h, l;
                split_bf16(af[q], h, l); ah[q] = h; al[q] = l;
                split_bf16(bf[q], h, l); bh[q] = h; bl[q] = l;
            }
            *(v4bf*)&a_hi[row][c4] = ah;
            *(v4bf*)&a_lo[row][c4] = al;
            *(v4bf*)&b_hi[row][c4] = bh;
            *(v4bf*)&b_lo[row][c4] = bl;
        }

        // Prefetch next K chunk while this one is consumed (L2-resident anyway).
        if (kc + KC < RDIM) {
            __builtin_prefetch(Abase + (size_t)(tid >> 1) * RDIM + kc + KC + (tid & 1) * 32, 0, 1);
            __builtin_prefetch(Bbase + (size_t)(tid >> 1) * RDIM + kc + KC + (tid & 1) * 32, 0, 1);
        }

        __syncthreads();

        // ---- compute: 3xBF16 compensated WMMA ----
#pragma unroll
        for (int kb = 0; kb < KC; kb += 32) {
#pragma unroll
            for (int i = 0; i < 4; ++i) {
                const int ar = rowbase + i * 16 + lr;
                const v16bf ah = load_frag(&a_hi[ar][0], kb, half);
                const v16bf al = load_frag(&a_lo[ar][0], kb, half);
#pragma unroll
                for (int j = 0; j < 2; ++j) {
                    const int bc = colbase + j * 16 + lr;
                    const v16bf bh = load_frag(&b_hi[bc][0], kb, half);
                    const v16bf bl = load_frag(&b_lo[bc][0], kb, half);
                    acc[i][j] = __builtin_amdgcn_wmma_f32_16x16x32_bf16(
                        false, ah, false, bh, (short)0, acc[i][j], false, false);
                    acc[i][j] = __builtin_amdgcn_wmma_f32_16x16x32_bf16(
                        false, ah, false, bl, (short)0, acc[i][j], false, false);
                    acc[i][j] = __builtin_amdgcn_wmma_f32_16x16x32_bf16(
                        false, al, false, bh, (short)0, acc[i][j], false, false);
                }
            }
        }

        __syncthreads();
    }

    // ---- epilogue: C/D layout -> coalesced f32 stores ----
    // lane 0-15 = cols 0-15, VGPR v holds row v (lanes<16) / row 8+v (lanes>=16)
    const size_t outE = (size_t)e * NROWS * ODIM;
#pragma unroll
    for (int i = 0; i < 4; ++i) {
#pragma unroll
        for (int j = 0; j < 2; ++j) {
            const int col = tileN * 128 + colbase + j * 16 + lr;
#pragma unroll
            for (int v = 0; v < 8; ++v) {
                const int row = tileM * 128 + rowbase + i * 16 + v + half * 8;
                out[outE + (size_t)row * ODIM + col] = acc[i][j][v];
            }
        }
    }
}

extern "C" void kernel_launch(void* const* d_in, const int* in_sizes, int n_in,
                              void* d_out, int out_size, void* d_ws, size_t ws_size,
                              hipStream_t stream) {
    const float* X = (const float*)d_in[0];   // [E, N, R]
    const float* W = (const float*)d_in[1];   // [E, OUT, R]
    float* out     = (float*)d_out;           // [E, N, OUT]

    dim3 grid(ODIM / 128, NROWS / 128, NUM_E);
    dim3 block(256, 1, 1);
    moe_grouped_gemm_bf16x3<<<grid, block, 0, stream>>>(X, W, out);
}
